// StandardAttention_20229295964662
// MI455X (gfx1250) — matrix operand order repaired and checked by reading, compiled, not run on hardware
//
#include <hip/hip_runtime.h>
#include <hip/hip_bf16.h>

typedef __attribute__((ext_vector_type(16))) __bf16 v16bf;
typedef __attribute__((ext_vector_type(8)))  float  v8f;
typedef __attribute__((ext_vector_type(4)))  int    v4i;
typedef v4i __attribute__((address_space(1))) gv4i;   // global
typedef v4i __attribute__((address_space(3))) lv4i;   // LDS

#define NHEAD 16
#define HDIM  128
#define TDIM  2048
#define HID   2048
#define ATT_SCALE 0.08838834764831845f   // 128^-0.5

#if defined(__gfx1250__) && __has_builtin(__builtin_amdgcn_global_load_async_to_lds_b128)
#define ASYNC_LDS 1
#else
#define ASYNC_LDS 0
#endif

__device__ __forceinline__ void wait_async0() {
#if ASYNC_LDS
#if __has_builtin(__builtin_amdgcn_s_wait_asynccnt)
    __builtin_amdgcn_s_wait_asynccnt(0);
#else
    asm volatile("s_wait_asynccnt 0" ::: "memory");
#endif
#endif
}

__device__ __forceinline__ void async_copy_b128(const __bf16* g, __bf16* l) {
#if ASYNC_LDS
    __builtin_amdgcn_global_load_async_to_lds_b128((gv4i*)g, (lv4i*)l, 0, 0);
#else
    *(uint4*)l = *(const uint4*)g;
#endif
}

union BF16Frag {
    v16bf v;
    uint4 q[2];
};

__device__ __forceinline__ __bf16 f2bf(float f) {
    return (__bf16)f;   // hardware f32->bf16 convert (RNE)
}

union BF4 { __bf16 b[4]; uint2 u; };

// ---------------------------------------------------------------------------
// fp32 -> bf16 cast, 4 elements per thread
// ---------------------------------------------------------------------------
__global__ void cast_f32_bf16_v4(const float4* __restrict__ src,
                                 __bf16* __restrict__ dst, int n4) {
    int i = blockIdx.x * blockDim.x + threadIdx.x;
    if (i >= n4) return;
    float4 f = src[i];
    BF4 o;
    o.b[0] = f2bf(f.x); o.b[1] = f2bf(f.y);
    o.b[2] = f2bf(f.z); o.b[3] = f2bf(f.w);
    ((uint2*)dst)[i] = o.u;
}

// ---------------------------------------------------------------------------
// C[M,N] = A[M,K] * W[N,K]^T + bias ; K = N = 2048, bf16 inputs, f32 accum
// block = 256 threads (8 waves), block tile 128x128, wave tile 64x32 (4x2 WMMA)
// ---------------------------------------------------------------------------
template<bool F32OUT>
__global__ __launch_bounds__(256)
void gemm_xwT(const __bf16* __restrict__ A,
              const __bf16* __restrict__ W,
              const float*  __restrict__ bias,
              void* __restrict__ Cout)
{
    const int K = 2048, N = 2048;
    __shared__ __align__(16) __bf16 sA[128 * 32];     // [m][k]
    __shared__ __align__(16) __bf16 sBt[32 * 136];    // [k][n], padded stride

    const int tid  = threadIdx.x;
    const int lane = tid & 31;
    const int wave = tid >> 5;
    const int wm   = wave & 1;     // 0..1  (64 rows each)
    const int wn   = wave >> 1;    // 0..3  (32 cols each)
    const int lh   = lane & 15;
    const int hi   = lane >> 4;
    const int m0   = blockIdx.y * 128;
    const int n0   = blockIdx.x * 128;

    v8f acc[4][2];
    const v8f vzero = {0.f,0.f,0.f,0.f,0.f,0.f,0.f,0.f};
    #pragma unroll
    for (int i = 0; i < 4; ++i)
        #pragma unroll
        for (int j = 0; j < 2; ++j) acc[i][j] = vzero;

    for (int k0 = 0; k0 < K; k0 += 32) {
        __syncthreads();
        // A tile: 128 rows x 32 halves = 512 b128 chunks, 2 per thread
        // (async DMA global->LDS on CDNA5)
        {
            int j = tid * 2;
            #pragma unroll
            for (int s = 0; s < 2; ++s, ++j) {
                int row = j >> 2;
                int c16 = (j & 3) * 8;
                async_copy_b128(A + (size_t)(m0 + row) * K + k0 + c16,
                                sA + row * 32 + c16);
            }
        }
        // W tile, stored transposed: sBt[k][n]
        {
            int j = tid * 2;
            #pragma unroll
            for (int s = 0; s < 2; ++s, ++j) {
                int row = j >> 2;                 // n within tile
                int c16 = (j & 3) * 8;            // k within tile
                uint4 d = *(const uint4*)(W + (size_t)(n0 + row) * K + k0 + c16);
                const __bf16* e = (const __bf16*)&d;
                #pragma unroll
                for (int q = 0; q < 8; ++q)
                    sBt[(c16 + q) * 136 + row] = e[q];
            }
        }
        // prefetch next k-step while this tile is consumed
        if (k0 + 32 < K) {
            int row = tid >> 1;
            __builtin_prefetch(A + (size_t)(m0 + row) * K + k0 + 32, 0, 1);
            __builtin_prefetch(W + (size_t)(n0 + row) * K + k0 + 32, 0, 1);
        }
        wait_async0();
        __syncthreads();

        BF16Frag af[4], bfr[2];
        #pragma unroll
        for (int i = 0; i < 4; ++i) {
            int mr = wm * 64 + i * 16 + lh;
            af[i].q[0] = *(const uint4*)(sA + mr * 32 + hi * 8);
            af[i].q[1] = *(const uint4*)(sA + mr * 32 + 16 + hi * 8);
        }
        #pragma unroll
        for (int j = 0; j < 2; ++j) {
            int nc = wn * 32 + j * 16;
            #pragma unroll
            for (int s = 0; s < 16; ++s)
                bfr[j].v[s] = sBt[((s & 8) * 2 + hi * 8 + (s & 7)) * 136 + nc + lh];
        }
        #pragma unroll
        for (int i = 0; i < 4; ++i)
            #pragma unroll
            for (int j = 0; j < 2; ++j)
                acc[i][j] = __builtin_amdgcn_wmma_f32_16x16x32_bf16(
                    false, af[i].v, false, bfr[j].v, (short)0, acc[i][j], false, false);
    }

    #pragma unroll
    for (int i = 0; i < 4; ++i) {
        #pragma unroll
        for (int j = 0; j < 2; ++j) {
            int col = n0 + wn * 32 + j * 16 + lh;
            float bv = bias[col];
            #pragma unroll
            for (int r = 0; r < 8; ++r) {
                int row = m0 + wm * 64 + i * 16 + r + 8 * hi;
                float v = acc[i][j][r] + bv;
                if (F32OUT) ((float*)Cout)[(size_t)row * N + col] = v;
                else        ((__bf16*)Cout)[(size_t)row * N + col] = f2bf(v);
            }
        }
    }
}

// ---------------------------------------------------------------------------
// Causal flash attention. grid = (T/128, B*NH), block = 256 (8 waves).
// Each wave owns 16 query rows; K/V tiles of 32 keys staged in LDS.
// ---------------------------------------------------------------------------
__global__ __launch_bounds__(256)
void flash_attn(const __bf16* __restrict__ Q,
                const __bf16* __restrict__ Kc,
                const __bf16* __restrict__ V,
                __bf16* __restrict__ O)
{
    const int tid  = threadIdx.x;
    const int lane = tid & 31;
    const int wave = tid >> 5;
    const int lh   = lane & 15;
    const int hi   = lane >> 4;

    const int head = blockIdx.y & (NHEAD - 1);
    const int b    = blockIdx.y >> 4;
    const int q0b  = blockIdx.x * 128;
    const int q0   = q0b + wave * 16;

    const size_t rowbase = (size_t)b * TDIM * HID + (size_t)head * HDIM;

    __shared__ __align__(16) __bf16 sKt[128 * 40];   // [d][key] transposed
    __shared__ __align__(16) __bf16 sV[32 * 136];    // [key][d]
    __shared__ __align__(16) __bf16 sP[8][16 * 32];  // per-wave P staging

    // Q fragments (16 rows x 128 d) straight from global
    BF16Frag qf[4];
    {
        const __bf16* qrow = Q + rowbase + (size_t)(q0 + lh) * HID;
        #pragma unroll
        for (int i = 0; i < 4; ++i) {
            qf[i].q[0] = *(const uint4*)(qrow + i * 32 + hi * 8);
            qf[i].q[1] = *(const uint4*)(qrow + i * 32 + 16 + hi * 8);
        }
    }

    const v8f vzero = {0.f,0.f,0.f,0.f,0.f,0.f,0.f,0.f};
    v8f ctx[8];
    #pragma unroll
    for (int dt = 0; dt < 8; ++dt) ctx[dt] = vzero;
    float mrow[8], lrow[8];
    #pragma unroll
    for (int r = 0; r < 8; ++r) { mrow[r] = -1e30f; lrow[r] = 0.f; }

    const int nkt = (q0b + 128) / 32;
    for (int kt = 0; kt < nkt; ++kt) {
        const int kk = kt * 32;
        __syncthreads();
        // V tile: straight copy -> async DMA to LDS
        {
            int j = tid * 2;
            #pragma unroll
            for (int s = 0; s < 2; ++s, ++j) {
                int krow = j >> 4;            // 0..31
                int c8   = (j & 15) * 8;      // 0..120
                async_copy_b128(V + rowbase + (size_t)(kk + krow) * HID + c8,
                                sV + krow * 136 + c8);
            }
        }
        // K tile: transpose through VGPRs into sKt[d][key]
        {
            int j = tid * 2;
            #pragma unroll
            for (int s = 0; s < 2; ++s, ++j) {
                int krow = j >> 4;
                int c8   = (j & 15) * 8;
                uint4 dk = *(const uint4*)(Kc + rowbase + (size_t)(kk + krow) * HID + c8);
                const __bf16* e = (const __bf16*)&dk;
                #pragma unroll
                for (int q = 0; q < 8; ++q) sKt[(c8 + q) * 40 + krow] = e[q];
            }
        }
        if (kt + 1 < nkt) {
            int krow = tid >> 3;              // 0..31
            __builtin_prefetch(Kc + rowbase + (size_t)(kk + 32 + krow) * HID, 0, 1);
            __builtin_prefetch(V  + rowbase + (size_t)(kk + 32 + krow) * HID, 0, 1);
        }
        wait_async0();
        __syncthreads();

        if (kk > q0 + 15) continue;   // tile fully masked for this wave (uniform per wave)

        // S = Q K^T for 32 keys (two 16-key subtiles)
        float pvals[2][8];
        float tmax[8];
        #pragma unroll
        for (int r = 0; r < 8; ++r) tmax[r] = -1e30f;

        #pragma unroll
        for (int sub = 0; sub < 2; ++sub) {
            v8f s = vzero;
            #pragma unroll
            for (int i = 0; i < 4; ++i) {
                BF16Frag kb;
                #pragma unroll
                for (int s2 = 0; s2 < 16; ++s2)
                    kb.v[s2] = sKt[(i * 32 + (s2 & 8) * 2 + hi * 8 + (s2 & 7)) * 40 + sub * 16 + lh];
                s = __builtin_amdgcn_wmma_f32_16x16x32_bf16(
                        false, qf[i].v, false, kb.v, (short)0, s, false, false);
            }
            int key = kk + sub * 16 + lh;
            #pragma unroll
            for (int r = 0; r < 8; ++r) {
                int row = q0 + r + 8 * hi;
                float v = s[r] * ATT_SCALE;
                if (key > row) v = -1e30f;
                pvals[sub][r] = v;
                tmax[r] = fmaxf(tmax[r], v);
            }
        }
        // online softmax (rows live in 16-lane half-groups)
        float alpha[8];
        #pragma unroll
        for (int r = 0; r < 8; ++r) {
            float t = tmax[r];
            t = fmaxf(t, __shfl_xor(t, 1, 32));
            t = fmaxf(t, __shfl_xor(t, 2, 32));
            t = fmaxf(t, __shfl_xor(t, 4, 32));
            t = fmaxf(t, __shfl_xor(t, 8, 32));
            float mn = fmaxf(mrow[r], t);
            alpha[r] = __expf(mrow[r] - mn);
            mrow[r]  = mn;
            float p0 = __expf(pvals[0][r] - mn);
            float p1 = __expf(pvals[1][r] - mn);
            pvals[0][r] = p0; pvals[1][r] = p1;
            float ts = p0 + p1;
            ts += __shfl_xor(ts, 1, 32);
            ts += __shfl_xor(ts, 2, 32);
            ts += __shfl_xor(ts, 4, 32);
            ts += __shfl_xor(ts, 8, 32);
            lrow[r] = lrow[r] * alpha[r] + ts;
        }
        #pragma unroll
        for (int dt = 0; dt < 8; ++dt)
            #pragma unroll
            for (int r = 0; r < 8; ++r) ctx[dt][r] *= alpha[r];

        // C-layout -> A-layout via per-wave LDS staging
        #pragma unroll
        for (int sub = 0; sub < 2; ++sub)
            #pragma unroll
            for (int r = 0; r < 8; ++r)
                sP[wave][(r + 8 * hi) * 32 + sub * 16 + lh] = f2bf(pvals[sub][r]);
        asm volatile("s_wait_dscnt 0" ::: "memory");
        BF16Frag pf;
        pf.q[0] = *(const uint4*)(&sP[wave][lh * 32 + hi * 8]);
        pf.q[1] = *(const uint4*)(&sP[wave][lh * 32 + 16 + hi * 8]);

        // ctx += P V
        #pragma unroll
        for (int dt = 0; dt < 8; ++dt) {
            BF16Frag vf;
            #pragma unroll
            for (int s2 = 0; s2 < 16; ++s2)
                vf.v[s2] = sV[((s2 & 8) * 2 + hi * 8 + (s2 & 7)) * 136 + dt * 16 + lh];
            ctx[dt] = __builtin_amdgcn_wmma_f32_16x16x32_bf16(
                          false, pf.v, false, vf.v, (short)0, ctx[dt], false, false);
        }
    }

    // finalize: divide by row sums, write bf16 ctx
    float inv[8];
    #pragma unroll
    for (int r = 0; r < 8; ++r) inv[r] = 1.0f / lrow[r];
    #pragma unroll
    for (int dt = 0; dt < 8; ++dt) {
        #pragma unroll
        for (int r = 0; r < 8; ++r) {
            int row = q0 + r + 8 * hi;
            int d   = dt * 16 + lh;
            O[rowbase + (size_t)row * HID + d] = f2bf(ctx[dt][r] * inv[r]);
        }
    }
}

// ---------------------------------------------------------------------------
extern "C" void kernel_launch(void* const* d_in, const int* in_sizes, int n_in,
                              void* d_out, int out_size, void* d_ws, size_t ws_size,
                              hipStream_t stream) {
    const float* x  = (const float*)d_in[0];
    const float* qw = (const float*)d_in[1];
    const float* qb = (const float*)d_in[2];
    const float* kw = (const float*)d_in[3];
    const float* kb = (const float*)d_in[4];
    const float* vw = (const float*)d_in[5];
    const float* vb = (const float*)d_in[6];
    const float* ow = (const float*)d_in[7];
    const float* ob = (const float*)d_in[8];
    float* out = (float*)d_out;

    __bf16* ws = (__bf16*)d_ws;
    const size_t XE = (size_t)8192 * 2048;      // x / q / k / v / ctx elems
    const size_t WE = (size_t)2048 * 2048;      // weight elems
    __bf16* xb  = ws;                 // later reused as ctx buffer
    __bf16* wqb = ws + XE;
    __bf16* wkb = wqb + WE;
    __bf16* wvb = wkb + WE;
    __bf16* wob = wvb + WE;
    __bf16* qbf = wob + WE;
    __bf16* kbf = qbf + XE;
    __bf16* vbf = kbf + XE;

    const int n4x = (int)(XE / 4);
    const int n4w = (int)(WE / 4);
    cast_f32_bf16_v4<<<(n4x + 255) / 256, 256, 0, stream>>>((const float4*)x,  xb,  n4x);
    cast_f32_bf16_v4<<<(n4w + 255) / 256, 256, 0, stream>>>((const float4*)qw, wqb, n4w);
    cast_f32_bf16_v4<<<(n4w + 255) / 256, 256, 0, stream>>>((const float4*)kw, wkb, n4w);
    cast_f32_bf16_v4<<<(n4w + 255) / 256, 256, 0, stream>>>((const float4*)vw, wvb, n4w);
    cast_f32_bf16_v4<<<(n4w + 255) / 256, 256, 0, stream>>>((const float4*)ow, wob, n4w);

    dim3 g(16, 64);   // N/128, M/128
    gemm_xwT<false><<<g, 256, 0, stream>>>(xb, wqb, qb, qbf);
    gemm_xwT<false><<<g, 256, 0, stream>>>(xb, wkb, kb, kbf);
    gemm_xwT<false><<<g, 256, 0, stream>>>(xb, wvb, vb, vbf);

    flash_attn<<<dim3(16, 64), 256, 0, stream>>>(qbf, kbf, vbf, xb);

    gemm_xwT<true><<<g, 256, 0, stream>>>(xb, wob, ob, out);
}
